// SAGELayer_59536836657512
// MI455X (gfx1250) — matrix operand order repaired and checked by reading, not compiled, not run on hardware
//
#include <hip/hip_runtime.h>

#define DIN  64
#define DE   64
#define DOUT 128
#define KM   128   // message GEMM K = DIN + DE
#define KA   192   // apply GEMM K = DIN + DOUT

#define MEDGES 64
#define MNODES 64
#define ASTR_M 136   // LDS stride (ushorts) for A tile, message kernel (128+8 pad)
#define ASTR_A 200   // LDS stride for A tile, apply kernel (192+8 pad)
#define BSTR   136   // LDS stride for B (weightT) tiles (128+8 pad)

typedef __attribute__((ext_vector_type(16))) __bf16 v16bf;
typedef __attribute__((ext_vector_type(8)))  __bf16 v8bf;
typedef __attribute__((ext_vector_type(8)))  float  v8f;

__device__ __forceinline__ unsigned short f32_to_bf16(float f) {
  union { float f; unsigned int u; } x; x.f = f;
  unsigned int u = x.u;
  unsigned int r = u + 0x7FFFu + ((u >> 16) & 1u);   // round-to-nearest-even
  if ((u & 0x7F800000u) == 0x7F800000u) r = u;       // keep inf/nan
  return (unsigned short)(r >> 16);
}

__device__ __forceinline__ float bf16_to_f32(unsigned short h) {
  union { unsigned int u; float f; } x;
  x.u = ((unsigned int)h) << 16;
  return x.f;
}

// Split-precision: f32 x  ->  hi = bf16(x), lo = bf16(x - hi).
// A*B ~= Ah*Bh + Ah*Bl + Al*Bh  recovers ~fp32 accuracy on matrix cores.
__device__ __forceinline__ void f32_split(float v, unsigned short& hi,
                                          unsigned short& lo) {
  unsigned short h = f32_to_bf16(v);
  hi = h;
  lo = f32_to_bf16(v - bf16_to_f32(h));
}

// A fragment: 16x32 bf16. lane L: row = caller-provided (mt*16 + (L&15)),
// K = k0 + 8*(L>=16) + {0..7, 16..23}
__device__ __forceinline__ v16bf load_a_frag(const unsigned short* lds, int row,
                                             int stride, int k0, int lane) {
  int kb = k0 + ((lane >> 4) << 3);
  v8bf lo = *reinterpret_cast<const v8bf*>(lds + row * stride + kb);
  v8bf hi = *reinterpret_cast<const v8bf*>(lds + row * stride + kb + 16);
  return __builtin_shufflevector(lo, hi, 0, 1, 2, 3, 4, 5, 6, 7,
                                 8, 9, 10, 11, 12, 13, 14, 15);
}

// B fragment: 32x16 bf16. lane L holds N = n0 + (L&15),
// K = k0 + 8*(L>=16) + {0..7, 16..23}
__device__ __forceinline__ v16bf load_b_frag(const unsigned short* lds, int k0,
                                             int stride, int n0, int lane) {
  const unsigned short* p = lds + (k0 + ((lane >> 4) << 3)) * stride + n0 + (lane & 15);
  unsigned short t[16];
#pragma unroll
  for (int s = 0; s < 16; ++s) t[s] = p[(s < 8 ? s : s + 8) * stride];
  v16bf r;
  __builtin_memcpy(&r, t, sizeof(r));
  return r;
}

__device__ __forceinline__ v8f wmma3(v16bf aH, v16bf aL, v16bf bH, v16bf bL,
                                     v8f acc) {
  acc = __builtin_amdgcn_wmma_f32_16x16x32_bf16(false, aH, false, bH, (short)0,
                                                acc, false, false);
  acc = __builtin_amdgcn_wmma_f32_16x16x32_bf16(false, aH, false, bL, (short)0,
                                                acc, false, false);
  acc = __builtin_amdgcn_wmma_f32_16x16x32_bf16(false, aL, false, bH, (short)0,
                                                acc, false, false);
  return acc;
}

__global__ void zero_kernel(float* p, long long n) {
  long long i = (long long)blockIdx.x * blockDim.x + threadIdx.x;
  long long stride = (long long)gridDim.x * blockDim.x;
  for (; i < n; i += stride) p[i] = 0.f;
}

__global__ void count_kernel(const int* __restrict__ dst,
                             float* __restrict__ cnt, int Ecount) {
  int i = blockIdx.x * blockDim.x + threadIdx.x;
  int stride = gridDim.x * blockDim.x;
  for (; i < Ecount; i += stride) unsafeAtomicAdd(&cnt[dst[i]], 1.0f);
}

// Message GEMM: m[e,:] = W_msg * cat(nfeats[src[e]], efeats[e]);
// scatter-add to seg[dst[e],:]
__global__ __launch_bounds__(256) void msg_kernel(
    const float* __restrict__ nfeats, const float* __restrict__ efeats,
    const int* __restrict__ src, const int* __restrict__ dst,
    const float* __restrict__ Wm, float* __restrict__ seg, int Ecount) {
  __shared__ unsigned short ldsAh[MEDGES * ASTR_M];
  __shared__ unsigned short ldsAl[MEDGES * ASTR_M];
  __shared__ unsigned short ldsBh[KM * BSTR];
  __shared__ unsigned short ldsBl[KM * BSTR];
  __shared__ int ldsDst[MEDGES];

  const int tid  = threadIdx.x;
  const int lane = tid & 31;
  const int wave = tid >> 5;
  const int ebase = blockIdx.x * MEDGES;

  // Warm L2 with the weight matrix (global_prefetch_b8).
  for (size_t off = (size_t)tid * 256; off < (size_t)DOUT * KM * sizeof(float);
       off += 256 * 256)
    __builtin_prefetch((const char*)Wm + off, 0, 1);

  // Stage W_msg transposed ([k][o]) as split bf16; global reads coalesced.
  for (int idx = tid; idx < DOUT * KM; idx += 256) {
    int o = idx >> 7;        // KM == 128
    int k = idx & (KM - 1);
    unsigned short h, l;
    f32_split(Wm[idx], h, l);
    ldsBh[k * BSTR + o] = h;
    ldsBl[k * BSTR + o] = l;
  }

  // Gather edge-input rows: 4 threads per edge, 32 features each
  {
    int el = tid >> 2;       // 0..63
    int part = tid & 3;      // 0..3
    int eg = ebase + el;
    unsigned short* oh = ldsAh + el * ASTR_M + part * 32;
    unsigned short* ol = ldsAl + el * ASTR_M + part * 32;
    if (eg < Ecount) {
      const float* srcp = (part < 2)
          ? nfeats + (long long)src[eg] * DIN + part * 32
          : efeats + (long long)eg * DE + (part - 2) * 32;
      const float4* p4 = reinterpret_cast<const float4*>(srcp);
#pragma unroll
      for (int i = 0; i < 8; ++i) {
        float4 v = p4[i];
        f32_split(v.x, oh[i * 4 + 0], ol[i * 4 + 0]);
        f32_split(v.y, oh[i * 4 + 1], ol[i * 4 + 1]);
        f32_split(v.z, oh[i * 4 + 2], ol[i * 4 + 2]);
        f32_split(v.w, oh[i * 4 + 3], ol[i * 4 + 3]);
      }
    } else {
      for (int i = 0; i < 32; ++i) { oh[i] = 0; ol[i] = 0; }
    }
  }
  if (tid < MEDGES) {
    int eg = ebase + tid;
    ldsDst[tid] = (eg < Ecount) ? dst[eg] : -1;
  }
  __syncthreads();

  const int n0 = wave * 16;   // 8 waves x 16 = 128 output columns
  v8f zero = {};
  v8f acc[4] = {zero, zero, zero, zero};

#pragma unroll
  for (int kt = 0; kt < 4; ++kt) {       // K = 128 = 4 x 32
    int k0 = kt * 32;
    v16bf bH = load_b_frag(ldsBh, k0, BSTR, n0, lane);
    v16bf bL = load_b_frag(ldsBl, k0, BSTR, n0, lane);
#pragma unroll
    for (int mt = 0; mt < 4; ++mt) {     // 64 edges = 4 x 16 rows
      int row = mt * 16 + (lane & 15);
      v16bf aH = load_a_frag(ldsAh, row, ASTR_M, k0, lane);
      v16bf aL = load_a_frag(ldsAl, row, ASTR_M, k0, lane);
      acc[mt] = wmma3(aH, aL, bH, bL, acc[mt]);
    }
  }

  // Scatter-add into segment sums (bias folded into apply stage).
  const int colL  = lane & 15;
  const int mHalf = (lane >> 4) << 3;    // lanes 16..31 hold M = 8..15
#pragma unroll
  for (int mt = 0; mt < 4; ++mt) {
#pragma unroll
    for (int r = 0; r < 8; ++r) {
      int el = mt * 16 + mHalf + r;
      int d = ldsDst[el];
      if (d >= 0)
        unsafeAtomicAdd(&seg[(long long)d * DOUT + n0 + colL], acc[mt][r]);
    }
  }
}

// Apply GEMM: out[n,:] = relu(W_apply * cat(nfeats[n], h_neigh[n]) + b_apply)
// h_neigh = cnt>0 ? seg/cnt + b_msg : 0
__global__ __launch_bounds__(256) void apply_kernel(
    const float* __restrict__ nfeats, const float* __restrict__ seg,
    const float* __restrict__ cnt, const float* __restrict__ Wa,
    const float* __restrict__ ba, const float* __restrict__ bm,
    float* __restrict__ out, int Ncount) {
  __shared__ unsigned short ldsAh[MNODES * ASTR_A];
  __shared__ unsigned short ldsAl[MNODES * ASTR_A];
  __shared__ unsigned short ldsBh[KA * BSTR];
  __shared__ unsigned short ldsBl[KA * BSTR];

  const int tid  = threadIdx.x;
  const int lane = tid & 31;
  const int wave = tid >> 5;
  const int nbase = blockIdx.x * MNODES;

  for (size_t off = (size_t)tid * 256; off < (size_t)DOUT * KA * sizeof(float);
       off += 256 * 256)
    __builtin_prefetch((const char*)Wa + off, 0, 1);

  for (int idx = tid; idx < DOUT * KA; idx += 256) {
    int o = idx / KA;
    int k = idx - o * KA;
    unsigned short h, l;
    f32_split(Wa[idx], h, l);
    ldsBh[k * BSTR + o] = h;
    ldsBl[k * BSTR + o] = l;
  }

  {
    int nl = tid >> 2;       // 0..63
    int part = tid & 3;      // 48 features each
    int ng = nbase + nl;
    unsigned short* oh = ldsAh + nl * ASTR_A + part * 48;
    unsigned short* ol = ldsAl + nl * ASTR_A + part * 48;
    if (ng < Ncount) {
      float c = cnt[ng];
      bool has = c > 0.f;
      float inv = has ? 1.f / c : 0.f;
      for (int i = 0; i < 48; ++i) {
        int f = part * 48 + i;
        float v;
        if (f < DIN) {
          v = nfeats[(long long)ng * DIN + f];
        } else {
          int j = f - DIN;
          v = has ? (seg[(long long)ng * DOUT + j] * inv + bm[j]) : 0.f;
        }
        f32_split(v, oh[i], ol[i]);
      }
    } else {
      for (int i = 0; i < 48; ++i) { oh[i] = 0; ol[i] = 0; }
    }
  }
  __syncthreads();

  const int n0 = wave * 16;
  v8f zero = {};
  v8f acc[4] = {zero, zero, zero, zero};

#pragma unroll
  for (int kt = 0; kt < 6; ++kt) {       // K = 192 = 6 x 32
    int k0 = kt * 32;
    v16bf bH = load_b_frag(ldsBh, k0, BSTR, n0, lane);
    v16bf bL = load_b_frag(ldsBl, k0, BSTR, n0, lane);
#pragma unroll
    for (int mt = 0; mt < 4; ++mt) {
      int row = mt * 16 + (lane & 15);
      v16bf aH = load_a_frag(ldsAh, row, ASTR_A, k0, lane);
      v16bf aL = load_a_frag(ldsAl, row, ASTR_A, k0, lane);
      acc[mt] = wmma3(aH, aL, bH, bL, acc[mt]);
    }
  }

  const int colL  = lane & 15;
  const int mHalf = (lane >> 4) << 3;
  float bias = ba[n0 + colL];
#pragma unroll
  for (int mt = 0; mt < 4; ++mt) {
#pragma unroll
    for (int r = 0; r < 8; ++r) {
      int ng = nbase + mt * 16 + mHalf + r;
      if (ng < Ncount) {
        float v = acc[mt][r] + bias;
        out[(long long)ng * DOUT + n0 + colL] = v > 0.f ? v : 0.f;
      }
    }
  }
}

extern "C" void kernel_launch(void* const* d_in, const int* in_sizes, int n_in,
                              void* d_out, int out_size, void* d_ws, size_t ws_size,
                              hipStream_t stream) {
  const float* nfeats = (const float*)d_in[0];
  const float* efeats = (const float*)d_in[1];
  const int*   src    = (const int*)d_in[2];
  const int*   dst    = (const int*)d_in[3];
  const float* Wm     = (const float*)d_in[4];
  const float* bm     = (const float*)d_in[5];
  const float* Wa     = (const float*)d_in[6];
  const float* ba     = (const float*)d_in[7];
  float* out = (float*)d_out;

  const int N = in_sizes[0] / DIN;
  const int E = in_sizes[2];

  float* seg = (float*)d_ws;                     // [N, DOUT] segment sums
  float* cnt = seg + (long long)N * DOUT;        // [N] degree counts

  long long zn = (long long)N * DOUT + N;
  zero_kernel<<<2048, 256, 0, stream>>>(seg, zn);
  count_kernel<<<2048, 256, 0, stream>>>(dst, cnt, E);

  int mgrid = (E + MEDGES - 1) / MEDGES;
  msg_kernel<<<mgrid, 256, 0, stream>>>(nfeats, efeats, src, dst, Wm, seg, E);

  int agrid = (N + MNODES - 1) / MNODES;
  apply_kernel<<<agrid, 256, 0, stream>>>(nfeats, seg, cnt, Wa, ba, bm, out, N);
}
